// AttentionPool2d_38543036514954
// MI455X (gfx1250) — matrix operand order repaired and checked by reading, not compiled, not run on hardware
//
#include <hip/hip_runtime.h>
#include <hip/hip_bf16.h>

#define EMBED 1024
#define HEADS 16
#define HD    64
#define HHW   16384     // 128*128
#define NQ    200
#define QP    208       // NQ padded to 16
#define LSEQ  16584     // NQ + HW
#define LP    16608     // LSEQ padded to 32

typedef __attribute__((ext_vector_type(16))) __bf16       v16bf;
typedef __attribute__((ext_vector_type(8)))  float        v8f;
typedef __attribute__((ext_vector_type(8)))  unsigned int v8u;

__device__ __forceinline__ unsigned short f2bf(float f) {
    unsigned int u = __float_as_uint(f);
    u += 0x7FFFu + ((u >> 16) & 1u);          // round-to-nearest-even
    return (unsigned short)(u >> 16);
}
__device__ __forceinline__ float bf2f(unsigned short h) {
    return __uint_as_float(((unsigned int)h) << 16);
}

__device__ __forceinline__ v8f wmma_bf16(v8u a8, v8u b8, v8f c) {
    v16bf a = __builtin_bit_cast(v16bf, a8);
    v16bf b = __builtin_bit_cast(v16bf, b8);
    return __builtin_amdgcn_wmma_f32_16x16x32_bf16(false, a, false, b,
                                                   (short)0, c, false, false);
}

// A-operand (16x32 bf16): one contiguous run of 16 halfwords per lane.
// p0 = base + row*stride + k0 + (lane>=16 ? 16 : 0)
__device__ __forceinline__ v8u load_A16(const unsigned short* p0) {
    uint4 x = *(const uint4*)(p0);
    uint4 y = *(const uint4*)(p0 + 8);
    v8u r;
    r[0]=x.x; r[1]=x.y; r[2]=x.z; r[3]=x.w;
    r[4]=y.x; r[5]=y.y; r[6]=y.z; r[7]=y.w;
    return r;
}

// B-operand (32x16 bf16), GEMM-K contiguous per lane.
// p0 = base + col*stride + k0 + (lane>=16 ? 16 : 0)
__device__ __forceinline__ v8u load_B16_contig(const unsigned short* p0) {
    uint4 x = *(const uint4*)(p0);
    uint4 y = *(const uint4*)(p0 + 8);
    v8u r;
    r[0]=x.x; r[1]=x.y; r[2]=x.z; r[3]=x.w;
    r[4]=y.x; r[5]=y.y; r[6]=y.z; r[7]=y.w;
    return r;
}

// Async global->LDS 16B transfer (CDNA5 GLOBAL_LOAD_ASYNC_TO_LDS_B128, GV mode).
__device__ __forceinline__ void async_b128(unsigned lds_off, const void* gaddr) {
    unsigned long long a = (unsigned long long)(size_t)gaddr;
    asm volatile("global_load_async_to_lds_b128 %0, %1, off"
                 :: "v"(lds_off), "v"(a) : "memory");
}

// ---------------- prep kernels ----------------

// LDS-tiled: writes xb [C,HW] and xs [HW,C] both coalesced.
__global__ void transpose_x_kernel(const float* __restrict__ x,
                                   unsigned short* __restrict__ xs,
                                   unsigned short* __restrict__ xb) {
    __shared__ unsigned short tile[32][33];
    const int LT = HHW / 32;                 // 512 tiles along l
    int l0 = (blockIdx.x % LT) * 32;
    int c0 = (blockIdx.x / LT) * 32;
    int tx = threadIdx.x & 31;
    int ty = threadIdx.x >> 5;               // 0..7
#pragma unroll
    for (int cc = ty; cc < 32; cc += 8) {
        unsigned short v = f2bf(x[(size_t)(c0 + cc) * HHW + l0 + tx]);
        xb[(size_t)(c0 + cc) * HHW + l0 + tx] = v;
        tile[cc][tx] = v;
    }
    __syncthreads();
#pragma unroll
    for (int ll = ty; ll < 32; ll += 8) {
        xs[(size_t)(l0 + ll) * EMBED + c0 + tx] = tile[tx][ll];
    }
}

__global__ void sigmoid_kernel(const float* __restrict__ masks,
                               unsigned short* __restrict__ mT,
                               float* __restrict__ winv) {
    int q = blockIdx.x;                 // 0..QP-1
    float acc = 0.f;
    for (int l = threadIdx.x; l < HHW; l += blockDim.x) {
        float v = 0.f;
        if (q < NQ) v = 1.f / (1.f + __expf(-masks[(size_t)q * HHW + l]));
        mT[(size_t)q * HHW + l] = f2bf(v);
        acc += v;
    }
    __shared__ float red[256];
    red[threadIdx.x] = acc;
    __syncthreads();
    for (int s = 128; s > 0; s >>= 1) {
        if ((int)threadIdx.x < s) red[threadIdx.x] += red[threadIdx.x + s];
        __syncthreads();
    }
    if (threadIdx.x == 0) winv[q] = 1.f / (red[0] + 0.001f);
}

__global__ void convert_bf16_kernel(const float* __restrict__ src,
                                    unsigned short* __restrict__ dst, int n) {
    int i = blockIdx.x * blockDim.x + threadIdx.x;
    if (i < n) dst[i] = f2bf(src[i]);
}

// ---------- pooling GEMM: mean[QP,C] = (mT @ xs) * winv, 16x64 per wave ----------

__global__ void pool_gemm_kernel(const unsigned short* __restrict__ mT,
                                 const unsigned short* __restrict__ xb,
                                 const float* __restrict__ winv,
                                 unsigned short* __restrict__ meanbf) {
    int wave = (blockIdx.x * blockDim.x + threadIdx.x) >> 5;
    int lane = threadIdx.x & 31;
    const int NT4 = EMBED / 64;         // 16
    int mt = wave / NT4, nt = wave % NT4;
    if (mt >= QP / 16) return;
    int hb = lane >> 4, l16 = lane & 15;
    int arow = mt * 16 + l16;
    int cb = nt * 64 + l16;
    const unsigned short* Arow = mT + (size_t)arow * HHW + hb * 16;
    const unsigned short* B0 = xb + (size_t)(cb +  0) * HHW + hb * 16;
    const unsigned short* B1 = xb + (size_t)(cb + 16) * HHW + hb * 16;
    const unsigned short* B2 = xb + (size_t)(cb + 32) * HHW + hb * 16;
    const unsigned short* B3 = xb + (size_t)(cb + 48) * HHW + hb * 16;
    v8f a0 = {}, a1 = {}, a2 = {}, a3 = {};
    // ping-pong software pipeline: two register sets, no rotation moves
    v8u aA  = load_A16(Arow);
    v8u b0A = load_B16_contig(B0);
    v8u b1A = load_B16_contig(B1);
    v8u b2A = load_B16_contig(B2);
    v8u b3A = load_B16_contig(B3);
    for (int k0 = 0; k0 < HHW; k0 += 64) {
        v8u aB  = load_A16(Arow + k0 + 32);
        v8u b0B = load_B16_contig(B0 + k0 + 32);
        v8u b1B = load_B16_contig(B1 + k0 + 32);
        v8u b2B = load_B16_contig(B2 + k0 + 32);
        v8u b3B = load_B16_contig(B3 + k0 + 32);
        a0 = wmma_bf16(aA, b0A, a0);
        a1 = wmma_bf16(aA, b1A, a1);
        a2 = wmma_bf16(aA, b2A, a2);
        a3 = wmma_bf16(aA, b3A, a3);
        aA  = load_A16(Arow + k0 + 64);          // last iter: benign over-read
        b0A = load_B16_contig(B0 + k0 + 64);
        b1A = load_B16_contig(B1 + k0 + 64);
        b2A = load_B16_contig(B2 + k0 + 64);
        b3A = load_B16_contig(B3 + k0 + 64);
        a0 = wmma_bf16(aB, b0B, a0);
        a1 = wmma_bf16(aB, b1B, a1);
        a2 = wmma_bf16(aB, b2B, a2);
        a3 = wmma_bf16(aB, b3B, a3);
    }
    v8f accs[4] = {a0, a1, a2, a3};
#pragma unroll
    for (int c = 0; c < 4; ++c) {
        int ncol = cb + 16 * c;
#pragma unroll
        for (int i = 0; i < 8; ++i) {
            int q = mt * 16 + i + 8 * hb;
            meanbf[(size_t)q * EMBED + ncol] = f2bf(accs[c][i] * winv[q]);
        }
    }
}

// ---------- projection GEMM: D = alpha*(A @ W^T + b), 16x64 per wave ----------
// A rows < split from A0, else A1 (clamped). Optional transposed bf16 output.

__global__ void proj_gemm_kernel(const unsigned short* __restrict__ A0,
                                 const unsigned short* __restrict__ A1,
                                 int split, int a1_rows, int Mt,
                                 const unsigned short* __restrict__ Wb,
                                 const float* __restrict__ bias, float alpha,
                                 unsigned short* __restrict__ outb,
                                 int ob_transposed, int ob_stride,
                                 float* __restrict__ outf, int outf_rows) {
    int wave = (blockIdx.x * blockDim.x + threadIdx.x) >> 5;
    int lane = threadIdx.x & 31;
    const int NT4 = EMBED / 64;         // 16
    int mt = wave / NT4, nt = wave % NT4;
    if (mt >= Mt) return;
    int hb = lane >> 4, l16 = lane & 15;
    int arow = mt * 16 + l16;
    const unsigned short* Arow;
    if (arow < split) {
        Arow = A0 + (size_t)arow * EMBED;
    } else {
        int r = arow - split;
        if (r >= a1_rows) r = 0;        // padded rows: garbage, masked downstream
        Arow = A1 + (size_t)r * EMBED;
    }
    Arow += hb * 16;
    int cb = nt * 64 + l16;
    const unsigned short* B0 = Wb + (size_t)(cb +  0) * EMBED + hb * 16;
    const unsigned short* B1 = Wb + (size_t)(cb + 16) * EMBED + hb * 16;
    const unsigned short* B2 = Wb + (size_t)(cb + 32) * EMBED + hb * 16;
    const unsigned short* B3 = Wb + (size_t)(cb + 48) * EMBED + hb * 16;
    v8f a0 = {}, a1 = {}, a2 = {}, a3 = {};
    v8u aA  = load_A16(Arow);
    v8u b0A = load_B16_contig(B0);
    v8u b1A = load_B16_contig(B1);
    v8u b2A = load_B16_contig(B2);
    v8u b3A = load_B16_contig(B3);
    for (int k0 = 0; k0 < EMBED; k0 += 64) {
        v8u aB  = load_A16(Arow + k0 + 32);
        v8u b0B = load_B16_contig(B0 + k0 + 32);
        v8u b1B = load_B16_contig(B1 + k0 + 32);
        v8u b2B = load_B16_contig(B2 + k0 + 32);
        v8u b3B = load_B16_contig(B3 + k0 + 32);
        a0 = wmma_bf16(aA, b0A, a0);
        a1 = wmma_bf16(aA, b1A, a1);
        a2 = wmma_bf16(aA, b2A, a2);
        a3 = wmma_bf16(aA, b3A, a3);
        aA  = load_A16(Arow + k0 + 64);          // last iter: benign over-read
        b0A = load_B16_contig(B0 + k0 + 64);
        b1A = load_B16_contig(B1 + k0 + 64);
        b2A = load_B16_contig(B2 + k0 + 64);
        b3A = load_B16_contig(B3 + k0 + 64);
        a0 = wmma_bf16(aB, b0B, a0);
        a1 = wmma_bf16(aB, b1B, a1);
        a2 = wmma_bf16(aB, b2B, a2);
        a3 = wmma_bf16(aB, b3B, a3);
    }
    v8f accs[4] = {a0, a1, a2, a3};
#pragma unroll
    for (int c = 0; c < 4; ++c) {
        int ncol = cb + 16 * c;
        float bv = bias[ncol];
#pragma unroll
        for (int i = 0; i < 8; ++i) {
            int row = mt * 16 + i + 8 * hb;
            float v = (accs[c][i] + bv) * alpha;
            if (outb) {
                size_t oi = ob_transposed ? (size_t)ncol * ob_stride + row
                                          : (size_t)row * EMBED + ncol;
                outb[oi] = f2bf(v);
            }
            if (outf && row < outf_rows) outf[(size_t)row * EMBED + ncol] = v;
        }
    }
}

// ---------------- flash attention over L with mask ----------------

__device__ __forceinline__ bool att_allowed(int q, int l, const unsigned short* mT) {
    if (l < NQ) return l == q;                       // pooled tokens: self only
    if (l >= LSEQ) return false;                     // padding
    return bf2f(mT[(size_t)q * HHW + (l - NQ)]) > 0.9f;
}

__global__ void attention_kernel(const unsigned short* __restrict__ qbf,
                                 const unsigned short* __restrict__ kbf,
                                 const unsigned short* __restrict__ vT,
                                 const unsigned short* __restrict__ mT,
                                 unsigned short* __restrict__ ctxbf) {
    __shared__ __align__(16) unsigned short Kst[8][2][32 * 64];  // 64 KB
    __shared__ __align__(16) unsigned short Pst[8][16 * 32];     //  8 KB
    int wv = threadIdx.x >> 5, lane = threadIdx.x & 31;
    int tile = blockIdx.x * 8 + wv;
    const int QT = QP / 16;              // 13
    int h = tile / QT, qt = tile % QT;
    if (h >= HEADS) return;
    int hb = lane >> 4, l16 = lane & 15;

    // Q tile (A-layout), K-dim = head dim, two 32-wide chunks; loop-invariant
    const unsigned short* qrow = qbf + (size_t)(qt * 16 + l16) * EMBED + h * HD + hb * 16;
    v8u qa0 = load_A16(qrow);
    v8u qa1 = load_A16(qrow + 32);

    float rm[8], rs[8];
    v8f o[4];
    v8f z = {};
#pragma unroll
    for (int i = 0; i < 8; ++i) { rm[i] = -1e30f; rs[i] = 0.f; }
    o[0] = z; o[1] = z; o[2] = z; o[3] = z;

    unsigned short* P = Pst[wv];

    // stage a 32x64 bf16 K tile into LDS asynchronously (8 b128 per lane)
    auto stage = [&](int l0, int buf) {
#pragma unroll
        for (int t = 0; t < 8; ++t) {
            int seg = t * 32 + lane;
            int row = seg >> 3;             // 0..31
            int sc  = (seg & 7) * 8;        // 0..56
            unsigned lds = (unsigned)(size_t)&Kst[wv][buf][row * 64 + sc];
            const unsigned short* g = kbf + (size_t)(l0 + row) * EMBED + h * HD + sc;
            async_b128(lds, g);
        }
    };
    stage(0, 0);

    for (int l0 = 0; l0 < LP; l0 += 32) {
        int buf = (l0 >> 5) & 1;
        bool more = (l0 + 32) < LP;
        if (more) {
            stage(l0 + 32, buf ^ 1);
            asm volatile("s_wait_asynccnt 8" ::: "memory");   // prev tile done
        } else {
            asm volatile("s_wait_asynccnt 0" ::: "memory");
        }

        // V-tile loads issued early: overlap with softmax VALU below
        v8u vb[4];
#pragma unroll
        for (int nb = 0; nb < 4; ++nb) {
            const unsigned short* vc =
                vT + (size_t)(h * HD + nb * 16 + l16) * LP + l0 + hb * 16;
            vb[nb] = load_B16_contig(vc);
        }

        // --- scores: two 16x16 tiles from LDS K, K = 64 as 2x32 ---
        const unsigned short* Kb = &Kst[wv][buf][0];
        const unsigned short* kc0 = Kb + (size_t)l16 * 64 + hb * 16;
        const unsigned short* kc1 = Kb + (size_t)(16 + l16) * 64 + hb * 16;
        v8f s0 = z, s1 = z;
        s0 = wmma_bf16(qa0, load_B16_contig(kc0), s0);
        s0 = wmma_bf16(qa1, load_B16_contig(kc0 + 32), s0);
        s1 = wmma_bf16(qa0, load_B16_contig(kc1), s1);
        s1 = wmma_bf16(qa1, load_B16_contig(kc1 + 32), s1);

        // --- mask + online softmax (rows across 16-lane halves) ---
        float p0[8], p1[8];
#pragma unroll
        for (int i = 0; i < 8; ++i) {
            int q  = qt * 16 + i + 8 * hb;
            float sA = att_allowed(q, l0 + l16,      mT) ? s0[i] : -1e30f;
            float sB = att_allowed(q, l0 + 16 + l16, mT) ? s1[i] : -1e30f;
            float t = fmaxf(sA, sB);
            t = fmaxf(t, __shfl_xor(t, 1));
            t = fmaxf(t, __shfl_xor(t, 2));
            t = fmaxf(t, __shfl_xor(t, 4));
            t = fmaxf(t, __shfl_xor(t, 8));
            float mn    = fmaxf(rm[i], t);
            float scale = __expf(rm[i] - mn);
            rm[i] = mn;
            float pa = __expf(sA - mn), pb = __expf(sB - mn);
            float ssum = pa + pb;
            ssum += __shfl_xor(ssum, 1);
            ssum += __shfl_xor(ssum, 2);
            ssum += __shfl_xor(ssum, 4);
            ssum += __shfl_xor(ssum, 8);
            rs[i] = rs[i] * scale + ssum;
#pragma unroll
            for (int nb = 0; nb < 4; ++nb) o[nb][i] *= scale;
            p0[i] = pa; p1[i] = pb;
        }

        // --- P: D-layout -> LDS -> A-layout ---
#pragma unroll
        for (int i = 0; i < 8; ++i) {
            int r = i + 8 * hb;
            P[r * 32 + l16]      = f2bf(p0[i]);
            P[r * 32 + 16 + l16] = f2bf(p1[i]);
        }
        asm volatile("s_wait_dscnt 0" ::: "memory");
        v8u pa = load_A16(P + l16 * 32 + hb * 16);

        // --- O += P @ V ---
#pragma unroll
        for (int nb = 0; nb < 4; ++nb)
            o[nb] = wmma_bf16(pa, vb[nb], o[nb]);
    }

#pragma unroll
    for (int i = 0; i < 8; ++i) {
        int q = qt * 16 + i + 8 * hb;
        if (q < NQ) {
            float inv = 1.0f / rs[i];
#pragma unroll
            for (int nb = 0; nb < 4; ++nb)
                ctxbf[(size_t)q * EMBED + h * HD + nb * 16 + l16] = f2bf(o[nb][i] * inv);
        }
    }
}

// ---------------- host launch ----------------

extern "C" void kernel_launch(void* const* d_in, const int* in_sizes, int n_in,
                              void* d_out, int out_size, void* d_ws, size_t ws_size,
                              hipStream_t stream) {
    const float* x     = (const float*)d_in[0];
    const float* masks = (const float*)d_in[1];
    const float* Wq    = (const float*)d_in[2];
    const float* bq    = (const float*)d_in[3];
    const float* Wk    = (const float*)d_in[4];
    const float* bk    = (const float*)d_in[5];
    const float* Wv    = (const float*)d_in[6];
    const float* bv    = (const float*)d_in[7];
    const float* Wc    = (const float*)d_in[8];
    const float* bc    = (const float*)d_in[9];

    char* ws = (char*)d_ws;
    size_t off = 0;
    auto alloc = [&](size_t bytes) {
        void* p = ws + off;
        off = (off + bytes + 255) & ~(size_t)255;
        return p;
    };
    unsigned short* xsb   = (unsigned short*)alloc((size_t)HHW * EMBED * 2);   // [HW,C]
    unsigned short* xbb   = (unsigned short*)alloc((size_t)EMBED * HHW * 2);   // [C,HW]
    unsigned short* mTb   = (unsigned short*)alloc((size_t)QP * HHW * 2);
    float*          wsum  = (float*)         alloc((size_t)QP * 4);
    unsigned short* wqb   = (unsigned short*)alloc((size_t)EMBED * EMBED * 2);
    unsigned short* wkb   = (unsigned short*)alloc((size_t)EMBED * EMBED * 2);
    unsigned short* wvb   = (unsigned short*)alloc((size_t)EMBED * EMBED * 2);
    unsigned short* wcb   = (unsigned short*)alloc((size_t)EMBED * EMBED * 2);
    unsigned short* meanb = (unsigned short*)alloc((size_t)QP * EMBED * 2);
    unsigned short* qb    = (unsigned short*)alloc((size_t)QP * EMBED * 2);
    unsigned short* kb    = (unsigned short*)alloc((size_t)LP * EMBED * 2);    // [LP,C]
    unsigned short* vtb   = (unsigned short*)alloc((size_t)EMBED * LP * 2);    // [C,LP]
    unsigned short* ctxb  = (unsigned short*)alloc((size_t)QP * EMBED * 2);
    (void)alloc(4096);   // tail pad: pipelined GEMMs over-read <=128B past buffers

    {
        int blocks = (HHW / 32) * (EMBED / 32);          // 16384
        transpose_x_kernel<<<blocks, 256, 0, stream>>>(x, xsb, xbb);
    }
    sigmoid_kernel<<<QP, 256, 0, stream>>>(masks, mTb, wsum);
    {
        int n = EMBED * EMBED, g = (n + 255) / 256;
        convert_bf16_kernel<<<g, 256, 0, stream>>>(Wq, wqb, n);
        convert_bf16_kernel<<<g, 256, 0, stream>>>(Wk, wkb, n);
        convert_bf16_kernel<<<g, 256, 0, stream>>>(Wv, wvb, n);
        convert_bf16_kernel<<<g, 256, 0, stream>>>(Wc, wcb, n);
    }

    // mask-weighted mean pooling
    {
        int waves = (QP / 16) * (EMBED / 64);            // 208
        pool_gemm_kernel<<<(waves + 7) / 8, 256, 0, stream>>>(mTb, xbb, wsum, meanb);
    }

    const float qscale = 0.125f;                         // hd^-0.5
    // Q projection
    {
        int Mt = QP / 16, waves = Mt * (EMBED / 64);
        proj_gemm_kernel<<<(waves + 7) / 8, 256, 0, stream>>>(
            meanb, meanb, QP, 1, Mt, wqb, bq, qscale, qb, 0, 0, (float*)nullptr, 0);
    }
    // K, V projections over seq = mean ++ xs (V written transposed [C, LP])
    {
        int Mt = LP / 16, waves = Mt * (EMBED / 64);
        proj_gemm_kernel<<<(waves + 7) / 8, 256, 0, stream>>>(
            meanb, xsb, NQ, HHW, Mt, wkb, bk, 1.0f, kb, 0, 0, (float*)nullptr, 0);
        proj_gemm_kernel<<<(waves + 7) / 8, 256, 0, stream>>>(
            meanb, xsb, NQ, HHW, Mt, wvb, bv, 1.0f, vtb, 1, LP, (float*)nullptr, 0);
    }
    // attention
    {
        int tiles = HEADS * (QP / 16);                   // 208
        attention_kernel<<<(tiles + 7) / 8, 256, 0, stream>>>(qb, kb, vtb, mTb, ctxb);
    }
    // output projection -> d_out (f32, rows < 200)
    {
        int Mt = QP / 16, waves = Mt * (EMBED / 64);
        proj_gemm_kernel<<<(waves + 7) / 8, 256, 0, stream>>>(
            ctxb, ctxb, QP, 1, Mt, wcb, bc, 1.0f,
            (unsigned short*)nullptr, 0, 0, (float*)d_out, NQ);
    }
}
